// Exl3HfLinear_74242804678962
// MI455X (gfx1250) — matrix operand order repaired and checked by reading, not compiled, not run on hardware
//
#include <hip/hip_runtime.h>
#include <stdint.h>

typedef __attribute__((ext_vector_type(16))) _Float16 v16h;
typedef __attribute__((ext_vector_type(8)))  _Float16 v8h;
typedef __attribute__((ext_vector_type(8)))  float    v8f;

#define K_FEAT   4096
#define N_FEAT   14336
#define KT_TILES 256      // K / 16
#define NT_TILES 896      // N / 16
#define KSPLIT   4
#define HADN     128
#define HNORM    0.08838834764831845f   // 1/sqrt(128)

// ---------------------------------------------------------------------------
// Kernel 1: A-prep.  axh[16][4096] f16 = Had128(x * suh) (rows 8..15 = 0).
// ---------------------------------------------------------------------------
__global__ __launch_bounds__(HADN) void exl3_xprep(
    const _Float16* __restrict__ x,
    const _Float16* __restrict__ suh,
    _Float16* __restrict__ axh) {
  __shared__ float s[HADN];
  const int t   = threadIdx.x;     // 0..127
  const int blk = blockIdx.x;      // 0..31
  const int row = blockIdx.y;      // 0..15 (8..15 are zero padding)
  const int col = blk * HADN + t;

  float v = 0.0f;
  if (row < 8) v = (float)x[row * K_FEAT + col] * (float)suh[col];
  s[t] = v;
  __syncthreads();

  #pragma unroll
  for (int st = 0; st < 7; ++st) {
    const int h = 1 << st;
    const float a = s[t];
    const float b = s[t ^ h];
    __syncthreads();
    s[t] = (t & h) ? (b - a) : (a + b);
    __syncthreads();
  }
  axh[row * K_FEAT + col] = (_Float16)(s[t] * HNORM);
}

// ---------------------------------------------------------------------------
// Kernel 2: trellis decode + WMMA GEMM, f32 partials per K-split.
// 8 waves / block; wave owns one 16-col N-tile.  Trellis tiles are staged
// through registers (software pipeline, 1 iteration ahead) and written to
// LDS with each 16-bit word expanded to a 32-bit slot, so the per-weight
// word pair (w, w+1) is two consecutive dwords off one per-lane base ->
// ds_load_2addr_b32.  One v_wmma_f32_16x16x32_f16 per K=32 step.
// ---------------------------------------------------------------------------
__global__ __launch_bounds__(256) void exl3_wmma_gemm(
    const _Float16* __restrict__ axh,       // [16][4096] f16
    const uint16_t* __restrict__ trellis,   // [256][896][48] u16
    float* __restrict__ yws) {              // [KSPLIT][8][N] f32 partials
  // expanded words: one u32 per 16-bit trellis word, +1 wrap word, pad to 56
  __shared__ uint32_t twx[8][2][56];

  const int tid  = threadIdx.x;
  const int wid  = tid >> 5;                // wave in workgroup (0..7)
  const int lane = tid & 31;
  const int nt   = blockIdx.x * 8 + wid;    // N-tile this wave computes
  const int kz   = blockIdx.y;              // K split (0..3)

  const int c      = lane & 15;             // column within N-tile
  const int tsel   = lane >> 4;             // which of the 2 k-tiles this lane decodes
  const uint32_t rsh = 16u - (uint32_t)((3 * c) & 15);  // circular window shift
  const int wbase  = (3 * c) >> 4;          // starting word (steps +3 per row)

  const uint32_t* tr32 = (const uint32_t*)trellis;
  const bool stager = lane < 24;

  v8f acc = {};
  const int kt0   = kz * (KT_TILES / KSPLIT);
  const int ktEnd = kt0 + (KT_TILES / KSPLIT);

  // ---- software pipeline: prime the first tile pair into registers ----
  uint32_t d0 = 0, d1 = 0;
  if (stager) {
    d0 = tr32[(kt0 * NT_TILES + nt) * 24 + lane];
    d1 = tr32[((kt0 + 1) * NT_TILES + nt) * 24 + lane];
  }

  #pragma unroll 2
  for (int kt = kt0; kt < ktEnd; kt += 2) {
    if (stager) {
      // expand current pair into LDS (u16 word w -> u32 slot w)
      const uint32_t lo0 = d0 & 0xFFFFu, hi0 = d0 >> 16;
      const uint32_t lo1 = d1 & 0xFFFFu, hi1 = d1 >> 16;
      twx[wid][0][2 * lane]     = lo0;      // word 2*lane
      twx[wid][0][2 * lane + 1] = hi0;      // word 2*lane+1
      twx[wid][1][2 * lane]     = lo1;
      twx[wid][1][2 * lane + 1] = hi1;
      if (lane == 0) {                      // tail-biting wrap word u[0]
        twx[wid][0][48] = lo0;
        twx[wid][1][48] = lo1;
      }
      // issue next pair's loads now; consumed next iteration
      const int ktn = (kt + 2 < ktEnd) ? (kt + 2) : kt0;
      const int idx = (ktn * NT_TILES + nt) * 24 + lane;
      d0 = tr32[idx];
      d1 = tr32[idx + NT_TILES * 24];
      // WGP-scope prefetch ~2 iterations ahead (constant byte offsets)
      __builtin_prefetch((const char*)&tr32[idx] + 2 * NT_TILES * 96, 0, 3);
      __builtin_prefetch((const char*)&tr32[idx] + 3 * NT_TILES * 96, 0, 3);
    }
    // wave-local LDS RAW: same-wave DS ops are processed in order.

    // ---- decode B fragment: 16 rows of column c of tile (kt + tsel) ----
    const uint32_t* tp = &twx[wid][tsel][wbase];
    v16h b;
    #pragma unroll
    for (int j = 0; j < 16; ++j) {
      const uint32_t hi = tp[3 * j];            // u[w]
      const uint32_t lo = tp[3 * j + 1];        // u[w+1]  (ds_load_2addr pair)
      const uint32_t state = (((hi << 16) | lo) >> rsh) & 0xFFFFu;
      uint32_t z = state * 89226354u + 64248484u;   // QTIP 3INST LCG
      z &= 0x8FFF8FFFu;
      const _Float16 hl = __builtin_bit_cast(_Float16, (unsigned short)(z & 0xFFFFu));
      const _Float16 hh = __builtin_bit_cast(_Float16, (unsigned short)(z >> 16));
      b[j] = hl + hh;
    }

    // ---- A fragment: row = c, contiguous 16 K per lane-half (matches B tile kt + tsel) ----
    const int kbase = kt * 16 + tsel * 16;
    const v8h a0 = *(const v8h*)(axh + c * K_FEAT + kbase);
    const v8h a1 = *(const v8h*)(axh + c * K_FEAT + kbase + 8);
    v16h a;
    #pragma unroll
    for (int j = 0; j < 8; ++j) { a[j] = a0[j]; a[8 + j] = a1[j]; }

    acc = __builtin_amdgcn_wmma_f32_16x16x32_f16(
        /*neg_a=*/false, a, /*neg_b=*/false, b,
        /*c_mod=*/(short)0, acc, /*reuse_a=*/false, /*reuse_b=*/false);
  }

  // C layout: lanes 0..15 hold rows M=0..7 in acc[0..7] — exactly our batch.
  if (lane < 16) {
    const int n = nt * 16 + c;
    #pragma unroll
    for (int m = 0; m < 8; ++m)
      yws[((size_t)(kz * 8 + m)) * N_FEAT + n] = acc[m];
  }
}

// ---------------------------------------------------------------------------
// Kernel 3: reduce K-splits, Hadamard-128 on output columns, scale+bias.
// ---------------------------------------------------------------------------
__global__ __launch_bounds__(HADN) void exl3_post(
    const float* __restrict__ yws,
    const _Float16* __restrict__ svh,
    const _Float16* __restrict__ bias,
    _Float16* __restrict__ out) {
  __shared__ float s[8][HADN];
  const int t = threadIdx.x;   // 0..127
  const int g = blockIdx.x;    // 0..111

  #pragma unroll
  for (int m = 0; m < 8; ++m) {
    float v = 0.0f;
    #pragma unroll
    for (int sp = 0; sp < KSPLIT; ++sp)
      v += yws[((size_t)(sp * 8 + m)) * N_FEAT + g * HADN + t];
    s[m][t] = v;
  }
  __syncthreads();

  for (int st = 0; st < 7; ++st) {
    const int h = 1 << st;
    for (int p = t; p < 512; p += HADN) {   // 512 pairs, each owned by 1 thread
      const int m  = p >> 6;
      const int pr = p & 63;
      const int i  = ((pr >> st) << (st + 1)) + (pr & (h - 1));
      const float a = s[m][i];
      const float b = s[m][i + h];
      s[m][i]     = a + b;
      s[m][i + h] = a - b;
    }
    __syncthreads();
  }

  const int n = g * HADN + t;
  const float sv = (float)svh[n];
  const float bs = (float)bias[n];
  #pragma unroll
  for (int m = 0; m < 8; ++m)
    out[(size_t)m * N_FEAT + n] = (_Float16)(s[m][t] * HNORM * sv + bs);
}

// ---------------------------------------------------------------------------
extern "C" void kernel_launch(void* const* d_in, const int* in_sizes, int n_in,
                              void* d_out, int out_size, void* d_ws, size_t ws_size,
                              hipStream_t stream) {
  (void)in_sizes; (void)n_in; (void)out_size; (void)ws_size;
  const _Float16* x    = (const _Float16*)d_in[0];
  const uint16_t* trel = (const uint16_t*)d_in[1];
  const _Float16* suh  = (const _Float16*)d_in[2];
  const _Float16* svh  = (const _Float16*)d_in[3];
  const _Float16* bias = (const _Float16*)d_in[4];
  _Float16* out = (_Float16*)d_out;

  _Float16* axh = (_Float16*)d_ws;                               // 128 KB
  float* yws = (float*)((char*)d_ws + (size_t)16 * K_FEAT * 2);  // ~1.75 MB

  exl3_xprep<<<dim3(K_FEAT / HADN, 16), HADN, 0, stream>>>(x, suh, axh);
  exl3_wmma_gemm<<<dim3(N_FEAT / HADN, KSPLIT), 256, 0, stream>>>(axh, trel, yws);
  exl3_post<<<dim3(N_FEAT / HADN), HADN, 0, stream>>>(yws, svh, bias, out);
}
